// Freq_TVT2_27015344292131
// MI455X (gfx1250) — matrix operand order repaired and checked by reading, not compiled, not run on hardware
//
#include <hip/hip_runtime.h>

// -----------------------------------------------------------------------------
// FNO spectral conv as a fused linear operator:
//   out[b,t,c] = sum_s input[b,s,c] * M[s,t]
// M is built on device from fno (irfft-basis GEMMs + DFT-basis GEMMs), then one
// big batched GEMM applies it. All GEMMs use one bf16x3 (hi/lo split) WMMA
// kernel: ~fp32 accuracy at bf16 matrix-unit rates (v_wmma_f32_16x16x32_bf16).
//
// Round 2: branch-free float4 fast-path tile loads (padded leading dims) +
// register double-buffering so next-tile global loads overlap the WMMAs.
// -----------------------------------------------------------------------------

typedef __attribute__((ext_vector_type(16))) __bf16 v16bf;
typedef __attribute__((ext_vector_type(8)))  float  v8f;

#define BLK_M 128
#define BLK_N 128
#define BLK_K 32
#define LDK   40   // LDS row stride in ushorts: 80 bytes, multiple of 16B for b128 loads

union FragU { v16bf v; uint4 q[2]; };

__device__ __forceinline__ void split_bf16(float x, unsigned short& hi, unsigned short& lo) {
  unsigned int b = __float_as_uint(x);
  unsigned int r = b + 0x7FFFu + ((b >> 16) & 1u);      // round-to-nearest-even bf16
  hi = (unsigned short)(r >> 16);
  float hf = __uint_as_float(((unsigned int)hi) << 16);
  float res = x - hf;                                    // residual
  unsigned int b2 = __float_as_uint(res);
  unsigned int r2 = b2 + 0x7FFFu + ((b2 >> 16) & 1u);
  lo = (unsigned short)(r2 >> 16);
}

// C = alpha * A@B + beta * C
// A: row-major [M x K], leading dim lda (contiguous in k).
// B: element B[k][n] at B[k*rsB + n*csB]  (supports transposed access).
// C: row-major [M x N], leading dim ldc.  Batching via blockIdx.z strides.
__global__ __launch_bounds__(256) void gemm_bf16x3(
    const float* __restrict__ A, const float* __restrict__ B, float* __restrict__ C,
    int M, int N, int K,
    int lda, long rsB, long csB, int ldc,
    float alpha, float beta,
    long batA, long batB, long batC)
{
  __shared__ unsigned short sAh[BLK_M * LDK];
  __shared__ unsigned short sAl[BLK_M * LDK];
  __shared__ unsigned short sBh[BLK_N * LDK];
  __shared__ unsigned short sBl[BLK_N * LDK];

  const int bm = blockIdx.x * BLK_M;
  const int bn = blockIdx.y * BLK_N;
  A += (long)blockIdx.z * batA;
  B += (long)blockIdx.z * batB;
  C += (long)blockIdx.z * batC;

  const int tid  = threadIdx.x;
  const int lane = tid & 31;
  const int wid  = tid >> 5;
  const int wm   = (wid & 3) * 32;   // 4 waves along M
  const int wn   = (wid >> 2) * 64;  // 2 waves along N
  const int lr   = lane & 15;        // row/col within 16x16 tile
  const int lh   = lane >> 4;        // K-half / M-half selector

  // per-thread tile-load mapping (float4 friendly)
  const int amRow = tid >> 3;        // 0..31  (+ i*32 -> 0..127)
  const int akCol = (tid & 7) * 4;   // 0,4,...,28
  const int bkRow = tid >> 5;        // 0..7   (+ i*8  -> 0..31)
  const int bnCol = (tid & 31) * 4;  // 0,4,...,124

  const bool alignA = ((lda & 3) == 0);
  const bool alignB = (csB == 1) && ((rsB & 3) == 0);
  const bool fullM  = (bm + BLK_M) <= M;
  const bool fullN  = (bn + BLK_N) <= N;

  float ra[4][4];
  float rb[4][4];

  auto loadTiles = [&](int kb) {
    const bool fullK = (kb + BLK_K) <= K;
    if (alignA && fullK && fullM) {
      #pragma unroll
      for (int i = 0; i < 4; ++i) {
        const float4 v = *(const float4*)&A[(long)(bm + amRow + i * 32) * lda + kb + akCol];
        ra[i][0] = v.x; ra[i][1] = v.y; ra[i][2] = v.z; ra[i][3] = v.w;
      }
    } else {
      #pragma unroll
      for (int i = 0; i < 4; ++i)
        #pragma unroll
        for (int j = 0; j < 4; ++j) {
          int gm = bm + amRow + i * 32, gk = kb + akCol + j;
          ra[i][j] = (gm < M && gk < K) ? A[(long)gm * lda + gk] : 0.0f;
        }
    }
    if (alignB && fullK && fullN) {
      #pragma unroll
      for (int i = 0; i < 4; ++i) {
        const float4 v = *(const float4*)&B[(long)(kb + bkRow + i * 8) * rsB + bn + bnCol];
        rb[i][0] = v.x; rb[i][1] = v.y; rb[i][2] = v.z; rb[i][3] = v.w;
      }
    } else {
      #pragma unroll
      for (int i = 0; i < 4; ++i)
        #pragma unroll
        for (int j = 0; j < 4; ++j) {
          int gk = kb + bkRow + i * 8, gn = bn + bnCol + j;
          rb[i][j] = (gk < K && gn < N) ? B[(long)gk * rsB + (long)gn * csB] : 0.0f;
        }
    }
  };

  auto stageToLds = [&]() {
    #pragma unroll
    for (int i = 0; i < 4; ++i) {
      int m = amRow + i * 32;
      #pragma unroll
      for (int j = 0; j < 4; ++j) {
        unsigned short h, l; split_bf16(ra[i][j], h, l);
        sAh[m * LDK + akCol + j] = h;
        sAl[m * LDK + akCol + j] = l;
      }
    }
    #pragma unroll
    for (int i = 0; i < 4; ++i) {
      int k = bkRow + i * 8;
      #pragma unroll
      for (int j = 0; j < 4; ++j) {
        unsigned short h, l; split_bf16(rb[i][j], h, l);
        sBh[(bnCol + j) * LDK + k] = h;   // stored transposed [n][k]
        sBl[(bnCol + j) * LDK + k] = l;
      }
    }
  };

  v8f acc[2][4];
  #pragma unroll
  for (int i = 0; i < 2; ++i)
    #pragma unroll
    for (int j = 0; j < 4; ++j)
      #pragma unroll
      for (int e = 0; e < 8; ++e) acc[i][j][e] = 0.0f;

  const int KT = (K + BLK_K - 1) / BLK_K;
  loadTiles(0);

  for (int kt = 0; kt < KT; ++kt) {
    stageToLds();
    __syncthreads();

    // register double-buffer: fetch next k-tile while this one computes
    if (kt + 1 < KT) loadTiles((kt + 1) * BLK_K);
    // L2 warm-up two tiles ahead (speculative global_prefetch)
    if (kt + 2 < KT) {
      int pk = (kt + 2) * BLK_K;
      __builtin_prefetch(&A[(long)(bm + amRow) * lda + pk + akCol], 0, 1);
      if (csB == 1) __builtin_prefetch(&B[(long)(pk + bkRow) * rsB + bn + bnCol], 0, 1);
    }

    // ---- build fragments per ISA 16-bit A (16x32) / B (32x16) layouts ----
    v16bf ah[2], al[2], bh[4], bl[4];
    #pragma unroll
    for (int mt = 0; mt < 2; ++mt) {
      const unsigned short* p = &sAh[(wm + mt * 16 + lr) * LDK];
      FragU f;
      f.q[0] = *(const uint4*)(p + lh * 8);        // K = h*8 .. h*8+7
      f.q[1] = *(const uint4*)(p + 16 + lh * 8);   // K = 16+h*8 ..
      ah[mt] = f.v;
      const unsigned short* p2 = &sAl[(wm + mt * 16 + lr) * LDK];
      f.q[0] = *(const uint4*)(p2 + lh * 8);
      f.q[1] = *(const uint4*)(p2 + 16 + lh * 8);
      al[mt] = f.v;
    }
    #pragma unroll
    for (int nt = 0; nt < 4; ++nt) {
      const unsigned short* p = &sBh[(wn + nt * 16 + lr) * LDK];
      FragU f;
      f.q[0] = *(const uint4*)(p + lh * 8);        // K = h*8 .. h*8+7
      f.q[1] = *(const uint4*)(p + 16 + lh * 8);   // K = 16+h*8 ..
      bh[nt] = f.v;
      const unsigned short* p2 = &sBl[(wn + nt * 16 + lr) * LDK];
      f.q[0] = *(const uint4*)(p2 + lh * 8);
      f.q[1] = *(const uint4*)(p2 + 16 + lh * 8);
      bl[nt] = f.v;
    }

    // ---- bf16x3: Ah*Bh + Ah*Bl + Al*Bh, fp32 accumulate ----
    #pragma unroll
    for (int mt = 0; mt < 2; ++mt)
      #pragma unroll
      for (int nt = 0; nt < 4; ++nt) {
        acc[mt][nt] = __builtin_amdgcn_wmma_f32_16x16x32_bf16(
            false, ah[mt], false, bh[nt], (short)0, acc[mt][nt], false, false);
        acc[mt][nt] = __builtin_amdgcn_wmma_f32_16x16x32_bf16(
            false, ah[mt], false, bl[nt], (short)0, acc[mt][nt], false, false);
        acc[mt][nt] = __builtin_amdgcn_wmma_f32_16x16x32_bf16(
            false, al[mt], false, bh[nt], (short)0, acc[mt][nt], false, false);
      }
    __syncthreads();
  }

  // ---- epilogue: C/D layout — lane = col, VGPR v = row v + 8*(lane/16) ----
  #pragma unroll
  for (int mt = 0; mt < 2; ++mt)
    #pragma unroll
    for (int nt = 0; nt < 4; ++nt) {
      int gn  = bn + wn + nt * 16 + lr;
      int gm0 = bm + wm + mt * 16 + lh * 8;
      if (gn < N) {
        #pragma unroll
        for (int v = 0; v < 8; ++v) {
          int gm = gm0 + v;
          if (gm < M) {
            long off = (long)gm * ldc + gn;
            float r = alpha * acc[mt][nt][v];
            if (beta != 0.0f) r += beta * C[off];
            C[off] = r;
          }
        }
      }
    }
}

// Cc[t,l] = (c_l/1024) cos(2*pi*l*t/1024), Cs likewise sin; c_l=1 at l in {0,512}, else 2.
// Stored with padded leading dim 528 (multiple of 4) for aligned GEMM loads.
__global__ __launch_bounds__(256) void gen_irfft_basis(float* __restrict__ Cc, float* __restrict__ Cs) {
  int idx = blockIdx.x * 256 + threadIdx.x;
  if (idx >= 1024 * 513) return;
  int t = idx / 513, l = idx - t * 513;
  float w = ((l == 0) || (l == 512)) ? (1.0f / 1024.0f) : (2.0f / 1024.0f);
  int ph = (l * t) & 1023;                     // exact phase reduction mod 1024
  float ang = (float)ph * (6.28318530717958647692f / 1024.0f);
  float s, c;
  __sincosf(ang, &s, &c);
  Cc[t * 528 + l] = w * c;
  Cs[t * 528 + l] = w * s;
}

// Cb[k, s-s0] = cos(2*pi*k*s/4096), Sb likewise sin; chunk of 1024 s-values.
__global__ __launch_bounds__(256) void gen_dft_basis(float* __restrict__ Cb, float* __restrict__ Sb, int s0) {
  int idx = blockIdx.x * 256 + threadIdx.x;
  if (idx >= 2049 * 1024) return;
  int k = idx >> 10, sc = idx & 1023;
  int s = s0 + sc;
  int ph = (k * s) & 4095;                     // exact phase reduction mod 4096
  float ang = (float)ph * (6.28318530717958647692f / 4096.0f);
  float sv, cv;
  __sincosf(ang, &sv, &cv);
  Cb[idx] = cv;
  Sb[idx] = sv;
}

extern "C" void kernel_launch(void* const* d_in, const int* in_sizes, int n_in,
                              void* d_out, int out_size, void* d_ws, size_t ws_size,
                              hipStream_t stream) {
  (void)in_sizes; (void)n_in; (void)out_size; (void)ws_size;
  const float* input  = (const float*)d_in[0];   // [32, 4096, 512]
  const float* fno_re = (const float*)d_in[1];   // [2049, 513]
  const float* fno_im = (const float*)d_in[2];   // [2049, 513]
  float* out = (float*)d_out;                    // [32, 1024, 512]

  // workspace layout (floats), ~54.8 MB total; padded leading dims (mult of 4)
  float* ws = (float*)d_ws;
  float* Cc = ws;                       // 1024*528
  float* Cs = Cc + 1024 * 528;          // 1024*528
  float* Pt = Cs + 1024 * 528;          // 1024*2064
  float* Qt = Pt + 1024 * 2064;         // 1024*2064
  float* Cb = Qt + 1024 * 2064;         // 2049*1024 (chunk)
  float* Sb = Cb + 2049 * 1024;         // 2049*1024 (chunk)
  float* Mt = Sb + 2049 * 1024;         // 1024*4096

  dim3 blk(256);

  // Stage 0: irfft basis
  gen_irfft_basis<<<(1024 * 513 + 255) / 256, 256, 0, stream>>>(Cc, Cs);

  // Stage A: Pt = Cc@Re' - Cs@Im' ; Qt = Cc@Im' + Cs@Re'   (X'[l,k] = X[k,l] -> rsB=1, csB=513)
  dim3 gA(1024 / 128, (2049 + 127) / 128, 1);
  gemm_bf16x3<<<gA, blk, 0, stream>>>(Cc, fno_re, Pt, 1024, 2049, 513, 528, 1L, 513L, 2064,  1.0f, 0.0f, 0, 0, 0);
  gemm_bf16x3<<<gA, blk, 0, stream>>>(Cs, fno_im, Pt, 1024, 2049, 513, 528, 1L, 513L, 2064, -1.0f, 1.0f, 0, 0, 0);
  gemm_bf16x3<<<gA, blk, 0, stream>>>(Cc, fno_im, Qt, 1024, 2049, 513, 528, 1L, 513L, 2064,  1.0f, 0.0f, 0, 0, 0);
  gemm_bf16x3<<<gA, blk, 0, stream>>>(Cs, fno_re, Qt, 1024, 2049, 513, 528, 1L, 513L, 2064,  1.0f, 1.0f, 0, 0, 0);

  // Stage B: Mt[t,s] = sum_k Pt[t,k] cos(2pi k s/4096) + Qt[t,k] sin(...), in 4 s-chunks
  dim3 gB(1024 / 128, 1024 / 128, 1);
  for (int c = 0; c < 4; ++c) {
    int s0 = c * 1024;
    gen_dft_basis<<<(2049 * 1024 + 255) / 256, 256, 0, stream>>>(Cb, Sb, s0);
    gemm_bf16x3<<<gB, blk, 0, stream>>>(Pt, Cb, Mt + s0, 1024, 1024, 2049, 2064, 1024L, 1L, 4096, 1.0f, 0.0f, 0, 0, 0);
    gemm_bf16x3<<<gB, blk, 0, stream>>>(Qt, Sb, Mt + s0, 1024, 1024, 2049, 2064, 1024L, 1L, 4096, 1.0f, 1.0f, 0, 0, 0);
  }

  // Stage C (dominant): out[b] = Mt[1024x4096] @ input[b][4096x512], batched over 32
  dim3 gC(1024 / 128, 512 / 128, 32);
  gemm_bf16x3<<<gC, blk, 0, stream>>>(Mt, input, out, 1024, 512, 4096, 4096, 512L, 1L, 512,
                                      1.0f, 0.0f, 0, 4096L * 512L, 1024L * 512L);
}
